// HeadAttention_11639361372683
// MI455X (gfx1250) — matrix operand order repaired and checked by reading, not compiled, not run on hardware
//
#include <hip/hip_runtime.h>
#include <hip/hip_bf16.h>

typedef _Float16 v4h  __attribute__((ext_vector_type(4)));
typedef _Float16 v8h  __attribute__((ext_vector_type(8)));
typedef _Float16 v16h __attribute__((ext_vector_type(16)));
typedef float    v4f  __attribute__((ext_vector_type(4)));
typedef float    v8f  __attribute__((ext_vector_type(8)));

#define N_EMBD 1024
#define HD     64
#define BATCH  8
#define SEQ    2048
#define NEGINF (-__builtin_inff())

#define QSZ ((size_t)BATCH * SEQ * HD)          // elements per Q/K/V plane

#define WMMA_F16(A,B,C) \
  __builtin_amdgcn_wmma_f32_16x16x32_f16(false,(A),false,(B),(short)0,(C),false,false)

// ---- CDNA5 async memory->LDS path (guarded; falls back to sync staging) ----
#if defined(__has_builtin)
#if __has_builtin(__builtin_amdgcn_global_load_async_to_lds_b128) && \
    __has_builtin(__builtin_amdgcn_s_wait_asynccnt)
#define USE_ASYNC_LDS 1
#endif
#endif

#ifdef USE_ASYNC_LDS
// builtin signature (from hipcc diagnostic): arg0 = int4 addrspace(1)*,
// arg1 = int4 addrspace(3)*, arg2 = imm offset, arg3 = imm cpol
typedef int v4i_g __attribute__((vector_size(16)));
#define GPTR(p) ((__attribute__((address_space(1))) v4i_g*)(p))
#define LPTR(p) ((__attribute__((address_space(3))) v4i_g*)(p))
#define ASYNC_CP_B128(dst_lds, src_glb) \
  __builtin_amdgcn_global_load_async_to_lds_b128(GPTR(src_glb), LPTR(dst_lds), 0, 0)
#endif

static __device__ __forceinline__ v16h cat8(v8h lo, v8h hi) {
  return __builtin_shufflevector(lo, hi, 0,1,2,3,4,5,6,7,8,9,10,11,12,13,14,15);
}

// xor-reductions across the 16 lanes of a half-wave (wave32, masks 1,2,4,8)
static __device__ __forceinline__ float redmax16(float v, unsigned lane) {
#pragma unroll
  for (int m = 1; m <= 8; m <<= 1) {
    int o = __builtin_amdgcn_ds_bpermute((int)((lane ^ (unsigned)m) << 2),
                                         __builtin_bit_cast(int, v));
    v = fmaxf(v, __builtin_bit_cast(float, o));
  }
  return v;
}
static __device__ __forceinline__ float redsum16(float v, unsigned lane) {
#pragma unroll
  for (int m = 1; m <= 8; m <<= 1) {
    int o = __builtin_amdgcn_ds_bpermute((int)((lane ^ (unsigned)m) << 2),
                                         __builtin_bit_cast(int, v));
    v += __builtin_bit_cast(float, o);
  }
  return v;
}

// ---------------------------------------------------------------------------
// Phase 0: one-shot f32 -> f16 weight repack: Wh[192][1024] = {Wq;Wk;Wv}.
// ---------------------------------------------------------------------------
__global__ __launch_bounds__(256)
void wconv_kernel(const float* __restrict__ Wq, const float* __restrict__ Wk,
                  const float* __restrict__ Wv, _Float16* __restrict__ Wh) {
  const int MS  = HD * N_EMBD;                        // 65536 (pow2)
  const int idx = ((int)blockIdx.x * 256 + (int)threadIdx.x) * 4;
  const float* src = (idx < MS) ? Wq : (idx < 2 * MS) ? Wk : Wv;
  v4f v = *(const v4f*)(src + (idx & (MS - 1)));
  v4h h;
#pragma unroll
  for (int j = 0; j < 4; ++j) h[j] = (_Float16)v[j];
  *(v4h*)(Wh + idx) = h;
}

// ---------------------------------------------------------------------------
// Phase 1: QKV projection.  One block = 16 rows of x, 128 threads (4 waves).
// x tile staged via global_load_async_to_lds_b128 (f32, converted at fragment
// load); B fragments are 2x b128 f16 loads from packed Wh.  Q,K stored
// row-major f16; V stored pre-transposed in 32-key tiles [B*S/32][64][32].
// ---------------------------------------------------------------------------
__global__ __launch_bounds__(128)
void qkv_proj_kernel(const float* __restrict__ x,
                     const _Float16* __restrict__ Wh,
                     const float* __restrict__ bq, const float* __restrict__ bk,
                     const float* __restrict__ bv,
                     _Float16* __restrict__ qkv) {
#ifdef USE_ASYNC_LDS
  __shared__ float xTf[16 * 40];                // f32 tile, stride 40 floats
#else
  __shared__ _Float16 xT[16 * 40];              // f16 tile, stride 40 halfs
#endif
  const unsigned t    = threadIdx.x;
  const unsigned lane = t & 31u;
  const unsigned w    = t >> 5;
  const unsigned half = lane >> 4;
  const unsigned l16  = lane & 15u;
  const int rowBase   = blockIdx.x * 16;

  v8f acc[3] = {};

  for (int e0 = 0; e0 < N_EMBD; e0 += 32) {
#ifdef USE_ASYNC_LDS
    {   // 16 rows x 128 B: one async b128 chunk per thread, memory -> LDS
      int r = (int)t >> 3, c = ((int)t & 7) * 4;
      ASYNC_CP_B128(&xTf[r * 40 + c], x + (size_t)(rowBase + r) * N_EMBD + e0 + c);
      __builtin_amdgcn_s_wait_asynccnt(0);
    }
#else
    {
      int idx = (int)t * 4;
      int r = idx >> 5, c = idx & 31;
      v4f xv = *(const v4f*)(x + (size_t)(rowBase + r) * N_EMBD + e0 + c);
      v4h hv;
#pragma unroll
      for (int j = 0; j < 4; ++j) hv[j] = (_Float16)xv[j];
      *(v4h*)&xT[r * 40 + c] = hv;
    }
#endif
    __syncthreads();

    // A fragment (16x32 f16): lane l holds row l%16, K per ISA layout
    const int kh = (int)half * 8;
    v16h afrag;
#ifdef USE_ASYNC_LDS
    {
      v8f f0 = *(const v8f*)&xTf[(int)l16 * 40 + kh];
      v8f f1 = *(const v8f*)&xTf[(int)l16 * 40 + 16 + kh];
#pragma unroll
      for (int j = 0; j < 8; ++j) { afrag[j] = (_Float16)f0[j]; afrag[8 + j] = (_Float16)f1[j]; }
    }
#else
    afrag = cat8(*(const v8h*)&xT[(int)l16 * 40 + kh],
                 *(const v8h*)&xT[(int)l16 * 40 + 16 + kh]);
#endif

#pragma unroll
    for (int ti = 0; ti < 3; ++ti) {
      const int tile = (int)w * 3 + ti;
      const _Float16* wp = Wh + ((size_t)tile * 16 + l16) * N_EMBD + e0 + (int)half * 8;
      v16h bfrag = cat8(*(const v8h*)wp, *(const v8h*)(wp + 16));
      acc[ti] = WMMA_F16(afrag, bfrag, acc[ti]);
    }
    __syncthreads();
  }

  // epilogue: +bias, f32 -> f16, store (V goes out pre-transposed)
#pragma unroll
  for (int ti = 0; ti < 3; ++ti) {
    const int tile = (int)w * 3 + ti;
    const int mat  = tile >> 2;
    const int col  = ((tile & 3) * 16) + (int)l16;
    const float* bias = (mat == 0) ? bq : (mat == 1) ? bk : bv;
    const float bvv = bias[col];
    if (mat < 2) {
      _Float16* outp = qkv + (size_t)mat * QSZ;
#pragma unroll
      for (int r = 0; r < 8; ++r) {
        int row = rowBase + r + (int)half * 8;
        outp[(size_t)row * HD + col] = (_Float16)(acc[ti][r] + bvv);
      }
    } else {
      _Float16* outp = qkv + 2 * QSZ;           // V^T tiles: [gr/32][64][32]
#pragma unroll
      for (int r = 0; r < 8; ++r) {
        int gr = rowBase + r + (int)half * 8;   // global row over B*S
        outp[((size_t)(gr >> 5) * HD + col) * 32 + (gr & 31)] = (_Float16)(acc[ti][r] + bvv);
      }
    }
  }
}

// ---------------------------------------------------------------------------
// Phase 2: flash attention.  One block = 64 query rows (4 waves x 16-row
// q-tile each).  K and pre-transposed V tiles staged with
// global_load_async_to_lds_b128 + s_wait_asynccnt.  Per 32-key tile:
// 4 WMMAs for S = Q@K^T, online softmax (ds_bpermute reductions), P restaged
// per-wave in LDS, 4 WMMAs for O += P@V.
// ---------------------------------------------------------------------------
__global__ __launch_bounds__(128)
void flash_attn_kernel(const _Float16* __restrict__ qkv,
                       const unsigned char* __restrict__ mask,  // [B][S] bool
                       float* __restrict__ out) {               // [B][S][64]
  __shared__ _Float16 kT[32 * 72];              // K tile row-major, stride 72
  __shared__ _Float16 vT[64 * 40];              // V^T tile [d][key], stride 40
  __shared__ _Float16 pst[4 * 16 * 40];         // per-wave P staging
  __shared__ __align__(16) unsigned char mRow[SEQ];

  const unsigned t    = threadIdx.x;
  const unsigned lane = t & 31u;
  const unsigned w    = t >> 5;
  const unsigned half = lane >> 4;
  const unsigned l16  = lane & 15u;
  const int b     = (int)blockIdx.x >> 5;
  const int qblk  = (int)blockIdx.x & 31;
  const int qBase = qblk * 64 + (int)w * 16;
  const int psB   = (int)w * 16 * 40;

  ((uint4*)mRow)[t] = ((const uint4*)(mask + (size_t)b * SEQ))[t];

  // Q A-fragments in registers for the whole loop (contiguous 16B chunks)
  const _Float16* qp = qkv + ((size_t)b * SEQ + qBase + l16) * HD;
  const int kh = (int)half * 8;
  v16h qa0 = cat8(*(const v8h*)(qp + kh),      *(const v8h*)(qp + 16 + kh));
  v16h qa1 = cat8(*(const v8h*)(qp + 32 + kh), *(const v8h*)(qp + 48 + kh));

  v8f accO[4] = {};
  float rowM[8], rowL[8];
#pragma unroll
  for (int r = 0; r < 8; ++r) { rowM[r] = NEGINF; rowL[r] = 0.0f; }

  __syncthreads();

  for (int kb = 0; kb < SEQ / 32; ++kb) {
    const _Float16* kbase = qkv + QSZ + ((size_t)b * SEQ + kb * 32) * HD;
    const _Float16* vtile = qkv + 2 * QSZ + (size_t)(b * (SEQ / 32) + kb) * (HD * 32);

#ifdef USE_ASYNC_LDS
    {   // K: 32 rows x 128 B  (2 async b128 per thread)
      int r = (int)t >> 2, c = ((int)t & 3) * 16;
      const _Float16* kp = kbase + (size_t)r * HD + c;
      ASYNC_CP_B128(&kT[r * 72 + c],     kp);
      ASYNC_CP_B128(&kT[r * 72 + c + 8], kp + 8);
    }
    {   // V^T: 64 rows x 64 B (2 async b128 per thread)
      int d = (int)t >> 1, c = ((int)t & 1) * 16;
      const _Float16* vp = vtile + d * 32 + c;
      ASYNC_CP_B128(&vT[d * 40 + c],     vp);
      ASYNC_CP_B128(&vT[d * 40 + c + 8], vp + 8);
    }
#else
    {
      int r = (int)t >> 2, c = ((int)t & 3) * 16;
      const _Float16* kp = kbase + (size_t)r * HD + c;
      *(v8h*)&kT[r * 72 + c]     = *(const v8h*)kp;
      *(v8h*)&kT[r * 72 + c + 8] = *(const v8h*)(kp + 8);
    }
    {
      int d = (int)t >> 1, c = ((int)t & 1) * 16;
      const _Float16* vp = vtile + d * 32 + c;
      *(v8h*)&vT[d * 40 + c]     = *(const v8h*)vp;
      *(v8h*)&vT[d * 40 + c + 8] = *(const v8h*)(vp + 8);
    }
#endif
    if (kb + 1 < SEQ / 32) {  // global_prefetch_b8 for next K tile
      __builtin_prefetch(kbase + 32 * HD + ((size_t)((int)t >> 2)) * HD, 0, 1);
    }
#ifdef USE_ASYNC_LDS
    __builtin_amdgcn_s_wait_asynccnt(0);
#endif
    __syncthreads();

    // --- S = Q @ K^T: two 16-key subtiles, K=64 as two k32 WMMAs each ---
    v8f s[2];
#pragma unroll
    for (int sub = 0; sub < 2; ++sub) {
      const int krow = sub * 16 + (int)l16;
      v16h b0 = cat8(*(const v8h*)&kT[krow * 72 + kh],
                     *(const v8h*)&kT[krow * 72 + 16 + kh]);
      v16h b1 = cat8(*(const v8h*)&kT[krow * 72 + 32 + kh],
                     *(const v8h*)&kT[krow * 72 + 48 + kh]);
      v8f z = {};
      z = WMMA_F16(qa0, b0, z);
      z = WMMA_F16(qa1, b1, z);
      s[sub] = z;
    }

    // --- mask + scale + online softmax ---
    const int key0 = kb * 32 + (int)l16;
    const bool m0 = mRow[key0] != 0;
    const bool m1 = mRow[key0 + 16] != 0;
#pragma unroll
    for (int r = 0; r < 8; ++r) {
      float s0 = m0 ? NEGINF : s[0][r] * 0.125f;
      float s1 = m1 ? NEGINF : s[1][r] * 0.125f;
      float mt   = redmax16(fmaxf(s0, s1), lane);
      float mNew = fmaxf(rowM[r], mt);
      float alpha = (mNew == rowM[r]) ? 1.0f : __expf(rowM[r] - mNew);
      float mexp  = (mNew == NEGINF) ? 0.0f : mNew;
      float p0 = __expf(s0 - mexp);
      float p1 = __expf(s1 - mexp);
      rowL[r] = rowL[r] * alpha + redsum16(p0 + p1, lane);
      rowM[r] = mNew;
#pragma unroll
      for (int nt = 0; nt < 4; ++nt) accO[nt][r] *= alpha;
      const int prow = r + (int)half * 8;
      pst[psB + prow * 40 + (int)l16]      = (_Float16)p0;
      pst[psB + prow * 40 + 16 + (int)l16] = (_Float16)p1;
    }

    asm volatile("" ::: "memory");  // per-wave DS ordering; block reordering

    // --- O += P @ V ---
    {
      const _Float16* pp = &pst[psB + (int)l16 * 40];
      v16h pf = cat8(*(const v8h*)(pp + kh), *(const v8h*)(pp + 16 + kh));
#pragma unroll
      for (int nt = 0; nt < 4; ++nt) {
        const int vrow = nt * 16 + (int)l16;
        v16h vf = cat8(*(const v8h*)&vT[vrow * 40 + kh],
                       *(const v8h*)&vT[vrow * 40 + 16 + kh]);
        accO[nt] = WMMA_F16(pf, vf, accO[nt]);
      }
    }
    __syncthreads();
  }

  // --- epilogue: normalize, write fp32 ---
#pragma unroll
  for (int nt = 0; nt < 4; ++nt) {
    const int col = nt * 16 + (int)l16;
#pragma unroll
    for (int r = 0; r < 8; ++r) {
      const int row = qBase + r + (int)half * 8;
      out[((size_t)b * SEQ + row) * HD + col] = accO[nt][r] / rowL[r];
    }
  }
}

extern "C" void kernel_launch(void* const* d_in, const int* in_sizes, int n_in,
                              void* d_out, int out_size, void* d_ws, size_t ws_size,
                              hipStream_t stream) {
  const float* x  = (const float*)d_in[0];
  const float* Wq = (const float*)d_in[1];
  const float* bq = (const float*)d_in[2];
  const float* Wk = (const float*)d_in[3];
  const float* bk = (const float*)d_in[4];
  const float* Wv = (const float*)d_in[5];
  const float* bv = (const float*)d_in[6];
  const unsigned char* mask = (const unsigned char*)d_in[7];

  _Float16* qkv = (_Float16*)d_ws;             // Q | K | V^T tiles | Wh  (~6.7 MB)
  _Float16* Wh  = qkv + 3 * QSZ;

  wconv_kernel<<<(3 * HD * N_EMBD) / (256 * 4), 256, 0, stream>>>(Wq, Wk, Wv, Wh);
  qkv_proj_kernel<<<(BATCH * SEQ) / 16, 128, 0, stream>>>(x, Wh, bq, bk, bv, qkv);
  flash_attn_kernel<<<BATCH * (SEQ / 64), 128, 0, stream>>>(qkv, mask, (float*)d_out);
}
